// AttentionBlock_30812095381994
// MI455X (gfx1250) — matrix operand order repaired and checked by reading, not compiled, not run on hardware
//
#include <hip/hip_runtime.h>
#include <hip/hip_bf16.h>
#include <math.h>

// Problem constants (from reference)
#define BATCH 4
#define CCH   256        // channels C
#define GRP   32         // groupnorm groups
#define CPG   8          // C / G
#define NTOK  4096       // H*W
#define DQKD  32         // q/k dim = C/8
// DQK^-0.5 * log2(e): softmax done in base-2 (v_exp_f32 is base-2 native)
#define QK_SCALE_LOG2E 0.25500299463927864f
#define EPSV 1e-5f

typedef _Float16 f16;
typedef _Float16 v16h __attribute__((ext_vector_type(16)));
typedef _Float16 v8h  __attribute__((ext_vector_type(8)));
typedef float    v8f  __attribute__((ext_vector_type(8)));

// ---------- WMMA operand loaders (wave32, 16x16x32 f16) ----------
// A matrix 16(M) x 32(K), source row-major with row stride ld (halfs).
// Lanes 0-15 row M=lane hold K {0..7,16..23}; lanes 16-31 row M=lane-16
// hold K {8..15,24..31}.
static __device__ __forceinline__ v16h load_A_f16(const f16* __restrict__ src, int ld) {
  const int lane = threadIdx.x & 31;
  const int row  = lane & 15;
  const int hh   = lane >> 4;
  const f16* p = src + row * ld + hh * 8;
  v8h lo = *(const v8h*)(p);       // K = 0..7   (+8 on upper half)
  v8h hi = *(const v8h*)(p + 16);  // K = 16..23 (+8 on upper half)
  v16h r;
#pragma unroll
  for (int i = 0; i < 8; ++i) { r[i] = lo[i]; r[8 + i] = hi[i]; }
  return r;
}

// B matrix 32(K) x 16(N), source column-major: each column's K values
// contiguous, column stride ld (halfs). Lanes 0-15: col=lane, K {0..7,16..23};
// lanes 16-31: col=lane-16, K {8..15,24..31} -> two 16-byte loads.
static __device__ __forceinline__ v16h load_B_f16(const f16* __restrict__ src, int ld) {
  const int lane = threadIdx.x & 31;
  const int col  = lane & 15;
  const int hh   = lane >> 4;
  const f16* p = src + (size_t)col * ld + hh * 8;
  v8h lo = *(const v8h*)(p);
  v8h hi = *(const v8h*)(p + 16);
  v16h r;
#pragma unroll
  for (int i = 0; i < 8; ++i) { r[i] = lo[i]; r[8 + i] = hi[i]; }
  return r;
}

static __device__ __forceinline__ v8f wmma_f16(v16h a, v16h b, v8f c) {
  return __builtin_amdgcn_wmma_f32_16x16x32_f16(false, a, false, b, (short)0, c, false, false);
}

// ---------- f32 -> f16 weight conversion ----------
__global__ void cvt_kernel(const float* __restrict__ s, f16* __restrict__ d, int n) {
  int i = blockIdx.x * 256 + threadIdx.x;
  if (i < n) d[i] = (f16)s[i];
}

// ---------- GroupNorm(32, 256) + affine, write h token-major [B][N][C] f16 ----------
__global__ void gn_kernel(const float* __restrict__ x, const float* __restrict__ gamma,
                          const float* __restrict__ beta, f16* __restrict__ ht) {
  const int b  = blockIdx.x / GRP;
  const int g  = blockIdx.x % GRP;
  const int c0 = g * CPG;
  const float* xp = x + ((size_t)b * CCH + c0) * NTOK;   // 8 contiguous channels
  __shared__ float red1[256], red2[256];
  const int M = CPG * NTOK;                              // 32768 elements
  float s = 0.f, sq = 0.f;
  for (int i = threadIdx.x; i < M; i += 256) { float v = xp[i]; s += v; sq += v * v; }
  red1[threadIdx.x] = s; red2[threadIdx.x] = sq;
  __syncthreads();
  for (int off = 128; off > 0; off >>= 1) {
    if (threadIdx.x < off) { red1[threadIdx.x] += red1[threadIdx.x + off];
                             red2[threadIdx.x] += red2[threadIdx.x + off]; }
    __syncthreads();
  }
  const float mu   = red1[0] / (float)M;
  const float var  = red2[0] / (float)M - mu * mu;
  const float rinv = rsqrtf(var + EPSV);
  for (int i = threadIdx.x; i < M; i += 256) {
    int cl = i >> 12;             // / 4096
    int n  = i & (NTOK - 1);
    int c  = c0 + cl;
    float v = (xp[i] - mu) * rinv * gamma[c] + beta[c];
    ht[((size_t)b * NTOK + n) * CCH + c] = (f16)v;       // transpose to token-major
  }
}

// ---------- generic WMMA GEMM: out[M,N] = alpha * (W[M,256] @ h[256,N]) + bias ----
// Register-blocked 16x64 strip per wave with an explicit software pipeline:
// next k-step's A + 4 B fragments are preloaded into distinct registers
// while the current step's 4 WMMAs execute, keeping ~10 b128 loads in
// flight past each WMMA instead of draining loadcnt to 0.
__global__ void gemm_qkv_kernel(const f16* __restrict__ w16, const float* __restrict__ bias,
                                const f16* __restrict__ ht, f16* __restrict__ out,
                                int M, float alpha, int token_major) {
  const int lane = threadIdx.x & 31;
  const int wave = threadIdx.x >> 5;
  const int tilesM = M >> 4;
  // grid: BATCH * tilesM * 8 blocks; each block = 8 waves * 4 n-tiles = 512 tokens
  const int b   = blockIdx.x / (tilesM * 8);
  const int rem = blockIdx.x % (tilesM * 8);
  const int mt  = rem / 8;
  const int nt0 = (rem % 8) * 32 + wave * 4;             // first of 4 n-tiles
  const f16* aBase = w16 + (size_t)mt * 16 * CCH;
  const f16* bBase = ht + ((size_t)b * NTOK + (size_t)nt0 * 16) * CCH;
  v8f acc[4] = {};
  v16h a  = load_A_f16(aBase, CCH);
  v16h b0 = load_B_f16(bBase + (size_t)0 * 16 * CCH, CCH);
  v16h b1 = load_B_f16(bBase + (size_t)1 * 16 * CCH, CCH);
  v16h b2 = load_B_f16(bBase + (size_t)2 * 16 * CCH, CCH);
  v16h b3 = load_B_f16(bBase + (size_t)3 * 16 * CCH, CCH);
#pragma unroll
  for (int k0 = 0; k0 < CCH; k0 += 32) {
    v16h an = {}, bn0 = {}, bn1 = {}, bn2 = {}, bn3 = {};
    if (k0 + 32 < CCH) {
      const int kn = k0 + 32;
      __builtin_prefetch(bBase + kn + 32, 0, 1);         // global_prefetch_b8
      an  = load_A_f16(aBase + kn, CCH);
      bn0 = load_B_f16(bBase + (size_t)0 * 16 * CCH + kn, CCH);
      bn1 = load_B_f16(bBase + (size_t)1 * 16 * CCH + kn, CCH);
      bn2 = load_B_f16(bBase + (size_t)2 * 16 * CCH + kn, CCH);
      bn3 = load_B_f16(bBase + (size_t)3 * 16 * CCH + kn, CCH);
    }
    acc[0] = wmma_f16(a, b0, acc[0]);
    acc[1] = wmma_f16(a, b1, acc[1]);
    acc[2] = wmma_f16(a, b2, acc[2]);
    acc[3] = wmma_f16(a, b3, acc[3]);
    a = an; b0 = bn0; b1 = bn1; b2 = bn2; b3 = bn3;
  }
  const int col = lane & 15, hh = lane >> 4;
#pragma unroll
  for (int j = 0; j < 4; ++j) {
#pragma unroll
    for (int r = 0; r < 8; ++r) {
      int m = mt * 16 + r + 8 * hh;                      // C/D layout row
      int n = (nt0 + j) * 16 + col;                      // C/D layout col
      float v = acc[j][r] * alpha + bias[m];
      if (token_major) out[((size_t)b * NTOK + n) * M + m] = (f16)v;
      else             out[((size_t)b * M + m) * NTOK + n] = (f16)v;
    }
  }
}

// ---------- fused flash attention: softmax(q kT) v ----------
// Block = 8 waves = 4 query tiles (64 queries) x 2 channel-halves (128 ch).
// All 8 V fragments + 2 K fragments are loaded at the TOP of each iteration,
// so their latency hides behind the S WMMAs and the softmax VALU sequence;
// once P returns from LDS the 8 PV WMMAs fire with data already resident.
// Softmax in base-2 (scale*log2e folded into q).
// q16,k16: [B][N][32] token-major; v16: [B][256][N]; o16: [B][N][256] f16.
__global__ void attn_kernel(const f16* __restrict__ q16, const f16* __restrict__ k16,
                            const f16* __restrict__ v16p, f16* __restrict__ o16) {
  __shared__ f16 plds[8 * 16 * 32];                      // per-wave P staging, 8 KB
  const int lane = threadIdx.x & 31;
  const int wave = threadIdx.x >> 5;
  const int qt   = wave >> 1;                            // query tile 0..3
  const int chh  = wave & 1;                             // channel half
  const int b    = blockIdx.x / (NTOK >> 6);
  const int n0   = (blockIdx.x % (NTOK >> 6)) * 64 + qt * 16;
  const int e0   = chh * 128;                            // this wave's 128 channels
  f16* pbuf = plds + wave * (16 * 32);

  const v16h qA = load_A_f16(q16 + ((size_t)b * NTOK + n0) * DQKD, DQKD);

  v8f oa[8] = {};
  float mrow[8], lrow[8];
#pragma unroll
  for (int r = 0; r < 8; ++r) { mrow[r] = -1e30f; lrow[r] = 0.f; }

  const int col = lane & 15, hh = lane >> 4;
  const f16* kBase = k16 + (size_t)b * NTOK * DQKD;
  const f16* vBase = v16p + (size_t)b * CCH * NTOK;

  for (int m0 = 0; m0 < NTOK; m0 += 32) {
    // prefetch next iteration's K tile
    __builtin_prefetch(kBase + (size_t)(m0 + 32) * DQKD, 0, 1);
    // ---- load phase: K fragments + all 8 V fragments for this key block
    v16h kb0 = load_B_f16(kBase + (size_t)m0 * DQKD, DQKD);
    v16h kb1 = load_B_f16(kBase + (size_t)(m0 + 16) * DQKD, DQKD);
    v16h vb0 = load_B_f16(vBase + (size_t)(e0 + 0 * 16) * NTOK + m0, NTOK);
    v16h vb1 = load_B_f16(vBase + (size_t)(e0 + 1 * 16) * NTOK + m0, NTOK);
    v16h vb2 = load_B_f16(vBase + (size_t)(e0 + 2 * 16) * NTOK + m0, NTOK);
    v16h vb3 = load_B_f16(vBase + (size_t)(e0 + 3 * 16) * NTOK + m0, NTOK);
    v16h vb4 = load_B_f16(vBase + (size_t)(e0 + 4 * 16) * NTOK + m0, NTOK);
    v16h vb5 = load_B_f16(vBase + (size_t)(e0 + 5 * 16) * NTOK + m0, NTOK);
    v16h vb6 = load_B_f16(vBase + (size_t)(e0 + 6 * 16) * NTOK + m0, NTOK);
    v16h vb7 = load_B_f16(vBase + (size_t)(e0 + 7 * 16) * NTOK + m0, NTOK);
    // ---- S tiles (A = Q 16x32, B = K^T 32x16, K-dim = d = 32)
    v8f zc = {};
    v8f s0 = wmma_f16(qA, kb0, zc);
    v8f s1 = wmma_f16(qA, kb1, zc);
    // ---- online softmax (base-2): row r of this lane is (r + 8*hh);
    // reductions stay inside 16-lane halves (xor 1,2,4,8) per C/D layout.
    // V-load latency hides behind this VALU block.
#pragma unroll
    for (int r = 0; r < 8; ++r) {
      float mx = fmaxf(s0[r], s1[r]);
      for (int off = 1; off <= 8; off <<= 1)
        mx = fmaxf(mx, __shfl_xor(mx, off, 32));
      float mnew = fmaxf(mrow[r], mx);
      float al   = exp2f(mrow[r] - mnew);
      float p0   = exp2f(s0[r] - mnew);
      float p1   = exp2f(s1[r] - mnew);
      float ps = p0 + p1;
      for (int off = 1; off <= 8; off <<= 1)
        ps += __shfl_xor(ps, off, 32);
      lrow[r] = lrow[r] * al + ps;
      mrow[r] = mnew;
#pragma unroll
      for (int t = 0; t < 8; ++t) oa[t][r] *= al;
      // stage P row-major 16x32 in LDS (lane owns one column per tile)
      pbuf[(r + 8 * hh) * 32 + col]      = (f16)p0;
      pbuf[(r + 8 * hh) * 32 + 16 + col] = (f16)p1;
    }
    // reload P in A-layout from LDS (same-wave DS ops are in-order)
    v16h pA = load_A_f16(pbuf, 32);
    // ---- 8 PV WMMAs, operands already resident
    oa[0] = wmma_f16(pA, vb0, oa[0]);
    oa[1] = wmma_f16(pA, vb1, oa[1]);
    oa[2] = wmma_f16(pA, vb2, oa[2]);
    oa[3] = wmma_f16(pA, vb3, oa[3]);
    oa[4] = wmma_f16(pA, vb4, oa[4]);
    oa[5] = wmma_f16(pA, vb5, oa[5]);
    oa[6] = wmma_f16(pA, vb6, oa[6]);
    oa[7] = wmma_f16(pA, vb7, oa[7]);
  }
#pragma unroll
  for (int r = 0; r < 8; ++r) {
    int n = n0 + r + 8 * hh;
    float inv = 1.f / lrow[r];
#pragma unroll
    for (int t = 0; t < 8; ++t)
      o16[((size_t)b * NTOK + n) * CCH + e0 + t * 16 + col] = (f16)(oa[t][r] * inv);
  }
}

// ---------- output projection + bias + residual (f32 out) ----------
// Same 16x64 register blocking / software pipeline as gemm_qkv.
__global__ void gemm_proj_kernel(const f16* __restrict__ wp16, const float* __restrict__ bp,
                                 const f16* __restrict__ o16, const float* __restrict__ x,
                                 float* __restrict__ out) {
  const int lane = threadIdx.x & 31;
  const int wave = threadIdx.x >> 5;
  // grid: BATCH * 16 * 8 blocks
  const int b   = blockIdx.x / (16 * 8);
  const int rem = blockIdx.x % (16 * 8);
  const int mt  = rem / 8;
  const int nt0 = (rem % 8) * 32 + wave * 4;
  const f16* aBase = wp16 + (size_t)mt * 16 * CCH;
  const f16* bBase = o16 + ((size_t)b * NTOK + (size_t)nt0 * 16) * CCH;
  v8f acc[4] = {};
  v16h a  = load_A_f16(aBase, CCH);
  v16h b0 = load_B_f16(bBase + (size_t)0 * 16 * CCH, CCH);
  v16h b1 = load_B_f16(bBase + (size_t)1 * 16 * CCH, CCH);
  v16h b2 = load_B_f16(bBase + (size_t)2 * 16 * CCH, CCH);
  v16h b3 = load_B_f16(bBase + (size_t)3 * 16 * CCH, CCH);
#pragma unroll
  for (int k0 = 0; k0 < CCH; k0 += 32) {
    v16h an = {}, bn0 = {}, bn1 = {}, bn2 = {}, bn3 = {};
    if (k0 + 32 < CCH) {
      const int kn = k0 + 32;
      __builtin_prefetch(bBase + kn + 32, 0, 1);
      an  = load_A_f16(aBase + kn, CCH);
      bn0 = load_B_f16(bBase + (size_t)0 * 16 * CCH + kn, CCH);
      bn1 = load_B_f16(bBase + (size_t)1 * 16 * CCH + kn, CCH);
      bn2 = load_B_f16(bBase + (size_t)2 * 16 * CCH + kn, CCH);
      bn3 = load_B_f16(bBase + (size_t)3 * 16 * CCH + kn, CCH);
    }
    acc[0] = wmma_f16(a, b0, acc[0]);
    acc[1] = wmma_f16(a, b1, acc[1]);
    acc[2] = wmma_f16(a, b2, acc[2]);
    acc[3] = wmma_f16(a, b3, acc[3]);
    a = an; b0 = bn0; b1 = bn1; b2 = bn2; b3 = bn3;
  }
  const int col = lane & 15, hh = lane >> 4;
#pragma unroll
  for (int j = 0; j < 4; ++j) {
#pragma unroll
    for (int r = 0; r < 8; ++r) {
      int f = mt * 16 + r + 8 * hh;
      int n = (nt0 + j) * 16 + col;
      size_t idx = ((size_t)b * CCH + f) * NTOK + n;
      out[idx] = x[idx] + acc[j][r] + bp[f];
    }
  }
}

extern "C" void kernel_launch(void* const* d_in, const int* in_sizes, int n_in,
                              void* d_out, int out_size, void* d_ws, size_t ws_size,
                              hipStream_t stream) {
  const float* x     = (const float*)d_in[0];
  const float* gamma = (const float*)d_in[1];
  const float* beta  = (const float*)d_in[2];
  const float* wq    = (const float*)d_in[3];
  const float* bq    = (const float*)d_in[4];
  const float* wk    = (const float*)d_in[5];
  const float* bk    = (const float*)d_in[6];
  const float* wv    = (const float*)d_in[7];
  const float* bv    = (const float*)d_in[8];
  const float* wp    = (const float*)d_in[9];
  const float* bp    = (const float*)d_in[10];
  (void)in_sizes; (void)n_in; (void)out_size; (void)ws_size;

  char* ws = (char*)d_ws;
  size_t off = 0;
  auto take = [&](size_t bytes) { char* p = ws + off; off += (bytes + 255) & ~(size_t)255; return p; };
  f16* ht   = (f16*)take((size_t)BATCH * NTOK * CCH * 2);   // h, token-major
  f16* q16  = (f16*)take((size_t)BATCH * NTOK * DQKD * 2);
  f16* k16  = (f16*)take((size_t)BATCH * NTOK * DQKD * 2);
  f16* v16b = (f16*)take((size_t)BATCH * CCH * NTOK * 2);   // v, channel-major
  f16* o16  = (f16*)take((size_t)BATCH * NTOK * CCH * 2);   // attn out, token-major
  f16* wq16 = (f16*)take((size_t)DQKD * CCH * 2);
  f16* wk16 = (f16*)take((size_t)DQKD * CCH * 2);
  f16* wv16 = (f16*)take((size_t)CCH * CCH * 2);
  f16* wp16 = (f16*)take((size_t)CCH * CCH * 2);

  cvt_kernel<<<(DQKD * CCH + 255) / 256, 256, 0, stream>>>(wq, wq16, DQKD * CCH);
  cvt_kernel<<<(DQKD * CCH + 255) / 256, 256, 0, stream>>>(wk, wk16, DQKD * CCH);
  cvt_kernel<<<(CCH * CCH + 255) / 256, 256, 0, stream>>>(wv, wv16, CCH * CCH);
  cvt_kernel<<<(CCH * CCH + 255) / 256, 256, 0, stream>>>(wp, wp16, CCH * CCH);

  gn_kernel<<<BATCH * GRP, 256, 0, stream>>>(x, gamma, beta, ht);

  // q (scale*log2e folded in), k : M=32 token-major; v : M=256 channel-major
  gemm_qkv_kernel<<<BATCH * 2 * 8, 256, 0, stream>>>(wq16, bq, ht, q16, DQKD, QK_SCALE_LOG2E, 1);
  gemm_qkv_kernel<<<BATCH * 2 * 8, 256, 0, stream>>>(wk16, bk, ht, k16, DQKD, 1.0f, 1);
  gemm_qkv_kernel<<<BATCH * 16 * 8, 256, 0, stream>>>(wv16, bv, ht, v16b, CCH, 1.0f, 0);

  attn_kernel<<<BATCH * (NTOK / 64), 256, 0, stream>>>(q16, k16, v16b, o16);

  gemm_proj_kernel<<<BATCH * 16 * 8, 256, 0, stream>>>(wp16, bp, o16, x, (float*)d_out);
}
